// DCNv2_Light_57878979281525
// MI455X (gfx1250) — matrix operand order repaired and checked by reading, not compiled, not run on hardware
//
#include <hip/hip_runtime.h>
#include <hip/hip_bf16.h>

// ---------------- constants (problem shape from the reference) -------------
#define Bn    4
#define Cin   64
#define Hc    128
#define Wc    128
#define Cout  128
#define KKf   9            // 3x3 taps
#define KTOT  576          // Cin * KK
#define NCH   27           // 18 offset + 9 mask channels
#define BSTR  592          // padded K stride for B tile in LDS (halves)

typedef __attribute__((ext_vector_type(16))) __bf16 v16bf;
typedef __attribute__((ext_vector_type(8)))  float  v8f;

// ---------------------------------------------------------------------------
// Kernel A: offset/mask 3x3 conv (27 output channels), sigmoid fused on mask.
// One thread per output pixel, all 27 channels in registers; weights in LDS.
// ---------------------------------------------------------------------------
__global__ __launch_bounds__(256) void offmask_conv(
    const float* __restrict__ x, const float* __restrict__ w_off,
    const float* __restrict__ b_off, const float* __restrict__ w_mask,
    const float* __restrict__ b_mask, float* __restrict__ offm)
{
    __shared__ float wS[NCH * KTOT];          // 27*576*4 = 62208 B
    const int tid = threadIdx.x;
    for (int i = tid; i < NCH * KTOT; i += 256) {
        int ch = i / KTOT, r = i - ch * KTOT;
        wS[i] = (ch < 18) ? w_off[ch * KTOT + r] : w_mask[(ch - 18) * KTOT + r];
    }
    __syncthreads();

    const int p   = blockIdx.x * 256 + tid;   // 65536 pixels total
    const int b   = p >> 14;
    const int rem = p & 16383;
    const int h   = rem >> 7;
    const int w   = rem & 127;

    float acc[NCH];
#pragma unroll
    for (int ch = 0; ch < 18; ++ch) acc[ch] = b_off[ch];
#pragma unroll
    for (int ch = 18; ch < NCH; ++ch) acc[ch] = b_mask[ch - 18];

    const float* xb = x + (size_t)b * Cin * Hc * Wc;
    for (int c = 0; c < Cin; ++c) {
        const float* xc = xb + c * (Hc * Wc);
#pragma unroll
        for (int ky = 0; ky < 3; ++ky) {
            const int yy = h - 1 + ky;
            if (yy < 0 || yy >= Hc) continue;
#pragma unroll
            for (int kx = 0; kx < 3; ++kx) {
                const int xx = w - 1 + kx;
                if (xx < 0 || xx >= Wc) continue;
                const float xv = xc[yy * Wc + xx];
                const int   wi = c * KKf + ky * 3 + kx;
#pragma unroll
                for (int ch = 0; ch < NCH; ++ch)
                    acc[ch] = fmaf(xv, wS[ch * KTOT + wi], acc[ch]);
            }
        }
    }
#pragma unroll
    for (int ch = 0; ch < NCH; ++ch) {
        float v = acc[ch];
        if (ch >= 18) v = 1.0f / (1.0f + __expf(-v));  // sigmoid for mask
        offm[(((size_t)b * NCH + ch) * Hc + h) * Wc + w] = v;
    }
}

// ---------------------------------------------------------------------------
// Kernel W: fp32 -> bf16 conversion of the (Cout x KTOT) GEMM weight matrix.
// ---------------------------------------------------------------------------
__global__ __launch_bounds__(256) void wcvt(const float* __restrict__ wgt,
                                            __bf16* __restrict__ wbf)
{
    const int i = blockIdx.x * 256 + threadIdx.x;
    if (i < Cout * KTOT) wbf[i] = (__bf16)wgt[i];
}

// ---------------------------------------------------------------------------
// Kernel B: deformable sampling + WMMA GEMM.
// One workgroup (128 threads / 4 waves) handles 16 adjacent pixels (one N
// tile) against all 128 output channels (8 M tiles, 2 per wave).
// ---------------------------------------------------------------------------
__global__ __launch_bounds__(128) void dcn_wmma(
    const float* __restrict__ x, const float* __restrict__ offm,
    const __bf16* __restrict__ wbf, float* __restrict__ out)
{
    __shared__ __bf16 Bs[16 * BSTR];   // sampled B tile, [n][k], 18944 B
    __shared__ int    sA[4][144];      // per (kk,n): 4 clamped corner addrs
    __shared__ float  sW[4][144];      // per (kk,n): 4 weights * mask * valid

    const int tid = threadIdx.x;
    const int p0  = blockIdx.x * 16;            // 16 pixels, same (b,h) row
    const int b   = p0 >> 14;
    const int rem = p0 & 16383;
    const int h   = rem >> 7;
    const int w0  = rem & 127;
    const int HW  = Hc * Wc;

    // ---- phase 1: bilinear sample parameters for 9 taps x 16 pixels ------
    for (int idx = tid; idx < 144; idx += 128) {
        const int kk = idx >> 4, n = idx & 15;
        const int ky = kk / 3,  kx = kk % 3;
        const int w  = w0 + n;
        const size_t obase = (((size_t)b * NCH) * Hc + h) * Wc + w;
        const float dy = offm[obase + (size_t)(2 * kk)     * HW];
        const float dx = offm[obase + (size_t)(2 * kk + 1) * HW];
        const float mk = offm[obase + (size_t)(18 + kk)    * HW];
        const float py = (float)(h - 1 + ky) + dy;
        const float px = (float)(w - 1 + kx) + dx;
        const float y0f = floorf(py), x0f = floorf(px);
        const float ly = py - y0f,   lx = px - x0f;
        const int   y0 = (int)y0f,   x0 = (int)x0f;
        const float cw[4] = { (1.f - ly) * (1.f - lx), (1.f - ly) * lx,
                              ly * (1.f - lx),          ly * lx };
#pragma unroll
        for (int j = 0; j < 4; ++j) {
            const int yi = y0 + (j >> 1), xi = x0 + (j & 1);
            const bool v = (yi >= 0) & (yi < Hc) & (xi >= 0) & (xi < Wc);
            const int yc = min(max(yi, 0), Hc - 1);
            const int xc = min(max(xi, 0), Wc - 1);
            sA[j][idx] = yc * Wc + xc;
            sW[j][idx] = v ? cw[j] * mk : 0.0f;   // clip-then-zero == reference
        }
    }
    __syncthreads();

    // ---- phase 2: build 576x16 sampled tile in bf16 (WMMA-B layout) ------
    const float* xb = x + (size_t)b * Cin * HW;
    for (int i = tid; i < KTOT * 16; i += 128) {
        const int n = i & 15, k = i >> 4;
        const int c = k / KKf, kk = k - c * KKf;
        const int s = kk * 16 + n;
        const float* xc = xb + c * HW;
        const float v = sW[0][s] * xc[sA[0][s]] + sW[1][s] * xc[sA[1][s]]
                      + sW[2][s] * xc[sA[2][s]] + sW[3][s] * xc[sA[3][s]];
        Bs[n * BSTR + k] = (__bf16)v;
    }
    __syncthreads();

    // ---- phase 3: 18-step K loop of v_wmma_f32_16x16x32_bf16 -------------
    const int wave = tid >> 5, lane = tid & 31;
    const int half = lane >> 4, ln = lane & 15;
    v8f c0 = {}, c1 = {};
    const int m0 = wave * 16 + ln;        // A row, M-tile #wave
    const int m1 = (wave + 4) * 16 + ln;  // A row, M-tile #wave+4
    // k order used here (contiguous, same for A and B; not the ISA table order):
    //  B elem j  <- B[K = kb*32 + half*16 + j][N = ln]          (contiguous 16)
    //  A elem j  <- A[m][K = kb*32 + half*16 + j]       (contiguous 16)
    const __bf16* bptr = &Bs[ln * BSTR + half * 16];
    const __bf16* a0p  = wbf + (size_t)m0 * KTOT + half * 16;
    const __bf16* a1p  = wbf + (size_t)m1 * KTOT + half * 16;
#pragma unroll 2
    for (int kb = 0; kb < 18; ++kb) {
        union { v16bf v; uint4 q[2]; } bu, au0, au1;
        bu.q[0]  = *(const uint4*)(bptr + kb * 32);       // K offs 0..7  (+half*16)
        bu.q[1]  = *(const uint4*)(bptr + kb * 32 + 8);   // K offs 8..15 (+half*16)
        au0.q[0] = *(const uint4*)(a0p + kb * 32);        // K offs 0..7  (+half*16)
        au0.q[1] = *(const uint4*)(a0p + kb * 32 + 8);    // K offs 8..15 (+half*16)
        au1.q[0] = *(const uint4*)(a1p + kb * 32);
        au1.q[1] = *(const uint4*)(a1p + kb * 32 + 8);
        c0 = __builtin_amdgcn_wmma_f32_16x16x32_bf16(false, au0.v, false, bu.v,
                                                     (short)0, c0, false, false);
        c1 = __builtin_amdgcn_wmma_f32_16x16x32_bf16(false, au1.v, false, bu.v,
                                                     (short)0, c1, false, false);
    }

    // ---- store: D elem r -> out[o = mtile*16 + half*8 + r][pixel ln] -----
    const size_t outb = ((size_t)b * Cout) * HW + (size_t)h * Wc + w0 + ln;
#pragma unroll
    for (int r = 0; r < 8; ++r) {
        const int o0 = wave * 16 + half * 8 + r;
        const int o1 = (wave + 4) * 16 + half * 8 + r;
        out[outb + (size_t)o0 * HW] = c0[r];
        out[outb + (size_t)o1 * HW] = c1[r];
    }
}

// ---------------------------------------------------------------------------
extern "C" void kernel_launch(void* const* d_in, const int* in_sizes, int n_in,
                              void* d_out, int out_size, void* d_ws, size_t ws_size,
                              hipStream_t stream)
{
    (void)in_sizes; (void)n_in; (void)out_size; (void)ws_size;
    const float* x      = (const float*)d_in[0];
    const float* w_off  = (const float*)d_in[1];
    const float* b_off  = (const float*)d_in[2];
    const float* w_mask = (const float*)d_in[3];
    const float* b_mask = (const float*)d_in[4];
    const float* weight = (const float*)d_in[5];
    float* out = (float*)d_out;

    // workspace layout: [ offm: 4*27*128*128 f32 | wbf: 128*576 bf16 ]
    float*  offm = (float*)d_ws;
    size_t  offm_bytes = (size_t)Bn * NCH * Hc * Wc * sizeof(float);
    __bf16* wbf = (__bf16*)((char*)d_ws + ((offm_bytes + 255) & ~(size_t)255));

    offmask_conv<<<(Bn * Hc * Wc) / 256, 256, 0, stream>>>(
        x, w_off, b_off, w_mask, b_mask, offm);
    wcvt<<<(Cout * KTOT + 255) / 256, 256, 0, stream>>>(weight, wbf);
    dcn_wmma<<<(Bn * Hc * Wc) / 16, 128, 0, stream>>>(x, offm, wbf, out);
}
